// Compute_Add_Attention_78735340470741
// MI455X (gfx1250) — matrix operand order repaired and checked by reading, not compiled, not run on hardware
//
#include <hip/hip_runtime.h>
#include <math.h>

typedef __attribute__((ext_vector_type(16))) _Float16 v16h;
typedef __attribute__((ext_vector_type(8)))  _Float16 v8h;
typedef __attribute__((ext_vector_type(4)))  _Float16 v4h;
typedef __attribute__((ext_vector_type(8)))  float    v8f;
typedef __attribute__((address_space(3)))    char     lds_char;

#define HID   1024
#define NCELL 32
#define BATCH 2048
#define N2H   2048                 // 2*H
#define MROWS (BATCH * NCELL)      // 65536

#define A_HALFS   (64 * 1024)      // A panel: 64 rows x 1024 f16
#define B_HALFS   (32 * 1024)      // one B buffer: 32 rows x 1024 f16
#define NT_PASSES (N2H / 32)       // 64

// ---------------------------------------------------------------------------
// CDNA5 async global->LDS helpers (ASYNCcnt-tracked, no VGPR round trip)
// ---------------------------------------------------------------------------
__device__ __forceinline__ void async_load_b128(unsigned lds_off, unsigned voff,
                                                const void* sbase) {
  asm volatile("global_load_async_to_lds_b128 %0, %1, %2 offset:0"
               :
               : "v"(lds_off), "v"(voff),
                 "s"((unsigned long long)(size_t)sbase)
               : "memory");
}
__device__ __forceinline__ void wait_asynccnt0() {
  asm volatile("s_wait_asynccnt 0" ::: "memory");
}
__device__ __forceinline__ unsigned lds_byte_offset(const void* p) {
  return (unsigned)(unsigned long long)(lds_char*)p;
}

// ---------------------------------------------------------------------------
// Kernel 1: W_in fp32 -> f16  (2048x1024)
// ---------------------------------------------------------------------------
__global__ __launch_bounds__(256)
void convert_w_kernel(const float* __restrict__ w, _Float16* __restrict__ wh) {
  int i = (blockIdx.x * 256 + threadIdx.x) * 4;
  float4 f = *(const float4*)(w + i);
  v4h h;
  h[0] = (_Float16)f.x; h[1] = (_Float16)f.y;
  h[2] = (_Float16)f.z; h[3] = (_Float16)f.w;
  *(v4h*)(wh + i) = h;
}

// ---------------------------------------------------------------------------
// Shared GEMM machinery: 128 threads = 4 waves, each wave one 16-row M-tile
// (full-K A panel in LDS as f16), two 16-col n-tiles per pass (shared A
// fragment, two independent acc chains). B panel is double-buffered and
// filled with global_load_async_to_lds_b128, overlapped with the WMMA loop.
// LDS: A 128KB + 2 x B 64KB = 256KB dynamic (<= 320KB WGP).
// ---------------------------------------------------------------------------

// issue 16 rows (n0..n0+15) of a B panel as 16 async b128 loads per thread
__device__ __forceinline__ void issue_b_panel_async(const _Float16* wh, int n0,
                                                    unsigned ldsB_bytes, int tid) {
#pragma unroll
  for (int it = 0; it < 16; ++it) {
    int g  = it * 128 + tid;
    int rr = g >> 7;                 // 0..15 row in panel
    int kc = (g & 127) << 3;         // half index within row
    unsigned voff = (unsigned)(((n0 + rr) << 10) + kc) * 2u;       // bytes
    unsigned lofs = ldsB_bytes + (unsigned)(rr * 1024 + kc) * 2u;  // bytes
    async_load_b128(lofs, voff, wh);
  }
}

// Kernel 2: proj_c = cell_state_c @ W_in^T   (M=2048, N=2048, K=1024)
__global__ __launch_bounds__(128)
void projc_kernel(const float* __restrict__ x,      // [2048,1024]
                  const _Float16* __restrict__ wh,  // [2048,1024]
                  float* __restrict__ projc)        // [2048,2048]
{
  extern __shared__ char smem[];
  _Float16* Ash = (_Float16*)smem;            // [64][1024]
  _Float16* Bsh = Ash + A_HALFS;              // [2][32][1024]
  const unsigned ldsBase = lds_byte_offset(smem);
  const unsigned ldsB    = ldsBase + A_HALFS * 2u;

  const int tid  = threadIdx.x;
  const int wave = tid >> 5;
  const int lane = tid & 31;
  const int half = lane >> 4;
  const int l15  = lane & 15;
  const int rowbase = blockIdx.x * 64;
  const int wrow    = rowbase + wave * 16;

  // overlap: start async fill of B buffer 0 before A staging
  issue_b_panel_async(wh, 0, ldsB, tid);

  // stage A (fp32 -> f16), coalesced float4 loads
  for (int it = 0; it < 128; ++it) {
    int g  = it * 128 + tid;
    int r  = g >> 8;
    int kc = (g & 255) << 2;
    float4 f = *(const float4*)(x + (size_t)(rowbase + r) * HID + kc);
    v4h h;
    h[0] = (_Float16)f.x; h[1] = (_Float16)f.y;
    h[2] = (_Float16)f.z; h[3] = (_Float16)f.w;
    *(v4h*)(Ash + r * 1024 + kc) = h;
  }
  wait_asynccnt0();
  __syncthreads();

  const int kbA = half * 8;
  const int koB = half * 8;
  const _Float16* Abase = Ash + (wave * 16 + l15) * 1024;

  for (int nt = 0; nt < NT_PASSES; ++nt) {
    const int n0  = nt * 32;
    const int cur = nt & 1;
    if (nt + 1 < NT_PASSES)
      issue_b_panel_async(wh, n0 + 32, ldsB + (unsigned)(1 - cur) * B_HALFS * 2u, tid);

    const _Float16* Bbuf   = Bsh + cur * B_HALFS;
    const _Float16* Bbase0 = Bbuf + l15 * 1024;
    const _Float16* Bbase1 = Bbuf + (16 + l15) * 1024;

    v8f acc0 = {};
    v8f acc1 = {};
#pragma unroll 4
    for (int k0 = 0; k0 < HID; k0 += 32) {
      v8h alo = *(const v8h*)(Abase + k0 + kbA);
      v8h ahi = *(const v8h*)(Abase + k0 + kbA + 16);
      v16h a  = __builtin_shufflevector(alo, ahi,
                  0,1,2,3,4,5,6,7,8,9,10,11,12,13,14,15);
      v8h b0l = *(const v8h*)(Bbase0 + k0 + koB);
      v8h b0h = *(const v8h*)(Bbase0 + k0 + koB + 16);
      v16h b0 = __builtin_shufflevector(b0l, b0h,
                  0,1,2,3,4,5,6,7,8,9,10,11,12,13,14,15);
      v8h b1l = *(const v8h*)(Bbase1 + k0 + koB);
      v8h b1h = *(const v8h*)(Bbase1 + k0 + koB + 16);
      v16h b1 = __builtin_shufflevector(b1l, b1h,
                  0,1,2,3,4,5,6,7,8,9,10,11,12,13,14,15);
      acc0 = __builtin_amdgcn_wmma_f32_16x16x32_f16(
                 false, a, false, b0, (short)0, acc0, false, false);
      acc1 = __builtin_amdgcn_wmma_f32_16x16x32_f16(
                 false, a, false, b1, (short)0, acc1, false, false);
    }
    // D layout: VGPR j -> M = j + 8*half, N = l15 (per tile)
#pragma unroll
    for (int j = 0; j < 8; ++j) {
      int gr = wrow + half * 8 + j;
      projc[(size_t)gr * N2H + n0 + l15]      = acc0[j];
      projc[(size_t)gr * N2H + n0 + 16 + l15] = acc1[j];
    }
    wait_asynccnt0();
    __syncthreads();
  }
}

// ---------------------------------------------------------------------------
// Kernel 3: fused  scores[b,c] = sum_n tanh(proj_c[b,n] + (cs@W^T)[b,c,n])*v[n]
// M = 65536 flat rows (b*32+c), N = 2048, K = 1024. proj_cells never hits HBM.
// ---------------------------------------------------------------------------
__global__ __launch_bounds__(128)
void scores_kernel(const float* __restrict__ cs,      // [65536,1024]
                   const _Float16* __restrict__ wh,   // [2048,1024]
                   const float* __restrict__ projc,   // [2048,2048]
                   const float* __restrict__ attv,    // [2048]
                   float* __restrict__ scores)        // [65536]
{
  extern __shared__ char smem[];
  _Float16* Ash = (_Float16*)smem;            // [64][1024]
  _Float16* Bsh = Ash + A_HALFS;              // [2][32][1024]
  const unsigned ldsBase = lds_byte_offset(smem);
  const unsigned ldsB    = ldsBase + A_HALFS * 2u;

  const int tid  = threadIdx.x;
  const int wave = tid >> 5;
  const int lane = tid & 31;
  const int half = lane >> 4;
  const int l15  = lane & 15;
  const int rowbase = blockIdx.x * 64;        // flat (b,c) row base
  const int wrow    = rowbase + wave * 16;    // 16-aligned -> single b
  const int b       = wrow >> 5;

  issue_b_panel_async(wh, 0, ldsB, tid);

  // stage A panel: 64 rows x 1024, fp32 -> f16
  for (int it = 0; it < 128; ++it) {
    int g  = it * 128 + tid;
    int r  = g >> 8;
    int kc = (g & 255) << 2;
    float4 f = *(const float4*)(cs + (size_t)(rowbase + r) * HID + kc);
    v4h h;
    h[0] = (_Float16)f.x; h[1] = (_Float16)f.y;
    h[2] = (_Float16)f.z; h[3] = (_Float16)f.w;
    *(v4h*)(Ash + r * 1024 + kc) = h;
  }
  wait_asynccnt0();
  __syncthreads();

  const int kbA = half * 8;
  const int koB = half * 8;
  const _Float16* Abase = Ash + (wave * 16 + l15) * 1024;

  float part[8] = {0.f, 0.f, 0.f, 0.f, 0.f, 0.f, 0.f, 0.f};

  for (int nt = 0; nt < NT_PASSES; ++nt) {
    const int n0  = nt * 32;
    const int cur = nt & 1;
    if (nt + 1 < NT_PASSES)
      issue_b_panel_async(wh, n0 + 32, ldsB + (unsigned)(1 - cur) * B_HALFS * 2u, tid);

    const _Float16* Bbuf   = Bsh + cur * B_HALFS;
    const _Float16* Bbase0 = Bbuf + l15 * 1024;
    const _Float16* Bbase1 = Bbuf + (16 + l15) * 1024;

    v8f acc0 = {};
    v8f acc1 = {};
#pragma unroll 4
    for (int k0 = 0; k0 < HID; k0 += 32) {
      v8h alo = *(const v8h*)(Abase + k0 + kbA);
      v8h ahi = *(const v8h*)(Abase + k0 + kbA + 16);
      v16h a  = __builtin_shufflevector(alo, ahi,
                  0,1,2,3,4,5,6,7,8,9,10,11,12,13,14,15);
      v8h b0l = *(const v8h*)(Bbase0 + k0 + koB);
      v8h b0h = *(const v8h*)(Bbase0 + k0 + koB + 16);
      v16h b0 = __builtin_shufflevector(b0l, b0h,
                  0,1,2,3,4,5,6,7,8,9,10,11,12,13,14,15);
      v8h b1l = *(const v8h*)(Bbase1 + k0 + koB);
      v8h b1h = *(const v8h*)(Bbase1 + k0 + koB + 16);
      v16h b1 = __builtin_shufflevector(b1l, b1h,
                  0,1,2,3,4,5,6,7,8,9,10,11,12,13,14,15);
      acc0 = __builtin_amdgcn_wmma_f32_16x16x32_f16(
                 false, a, false, b0, (short)0, acc0, false, false);
      acc1 = __builtin_amdgcn_wmma_f32_16x16x32_f16(
                 false, a, false, b1, (short)0, acc1, false, false);
    }
    // epilogue: tanh(proj_c + P) * v, per-lane N columns n0+l15, n0+16+l15
    float pc0 = projc[(size_t)b * N2H + n0 + l15];
    float pc1 = projc[(size_t)b * N2H + n0 + 16 + l15];
    float vv0 = attv[n0 + l15];
    float vv1 = attv[n0 + 16 + l15];
#pragma unroll
    for (int j = 0; j < 8; ++j) {
      part[j] += tanhf(pc0 + acc0[j]) * vv0;
      part[j] += tanhf(pc1 + acc1[j]) * vv1;
    }
    wait_asynccnt0();
    __syncthreads();
  }

  // reduce over the 16 lanes of each half (xor 1,2,4,8 stays in-group)
#pragma unroll
  for (int j = 0; j < 8; ++j) {
    float s = part[j];
    s += __shfl_xor(s, 1);
    s += __shfl_xor(s, 2);
    s += __shfl_xor(s, 4);
    s += __shfl_xor(s, 8);
    part[j] = s;
  }
  if (l15 == 0) {
#pragma unroll
    for (int j = 0; j < 8; ++j)
      scores[wrow + half * 8 + j] = part[j];   // half=0 -> M0..7, half=1 -> M8..15
  }
}

// ---------------------------------------------------------------------------
// Kernel 4: softmax over C=32 + atten_output (bandwidth bound, coalesced)
// ---------------------------------------------------------------------------
__global__ __launch_bounds__(256)
void softmax_out_kernel(const float* __restrict__ scores,  // [2048][32]
                        const float* __restrict__ cs,      // [2048][32][1024]
                        float* __restrict__ out_att,       // [2048][1024]
                        float* __restrict__ out_probs)     // [2048][32]
{
  const int b   = blockIdx.x;
  const int tid = threadIdx.x;
  float s[NCELL];
  float m = -INFINITY;
#pragma unroll
  for (int c = 0; c < NCELL; ++c) {
    s[c] = scores[b * NCELL + c];
    m = fmaxf(m, s[c]);
  }
  float sum = 0.f;
#pragma unroll
  for (int c = 0; c < NCELL; ++c) { s[c] = expf(s[c] - m); sum += s[c]; }
  const float inv = 1.f / sum;
#pragma unroll
  for (int c = 0; c < NCELL; ++c) s[c] *= inv;

  if (tid < NCELL) out_probs[b * NCELL + tid] = s[tid];

  for (int hh = tid; hh < HID; hh += 256) {
    float a = 0.f;
#pragma unroll
    for (int c = 0; c < NCELL; ++c)
      a += s[c] * cs[(size_t)b * (NCELL * HID) + c * HID + hh];
    out_att[(size_t)b * HID + hh] = a;
  }
}

// ---------------------------------------------------------------------------
extern "C" void kernel_launch(void* const* d_in, const int* in_sizes, int n_in,
                              void* d_out, int out_size, void* d_ws, size_t ws_size,
                              hipStream_t stream) {
  const float* cell_state_c = (const float*)d_in[0];   // [B,H]
  const float* cell_states  = (const float*)d_in[1];   // [B,C,H]
  const float* W_in         = (const float*)d_in[2];   // [2H,H]
  const float* atten_v      = (const float*)d_in[3];   // [2H]

  char* ws = (char*)d_ws;
  _Float16* Wh  = (_Float16*)ws;                                // 4 MB
  float* projc  = (float*)(ws + (size_t)4  * 1024 * 1024);      // 16 MB
  float* scores = (float*)(ws + (size_t)20 * 1024 * 1024);      // 256 KB

  float* out_att   = (float*)d_out;                 // [B,H]
  float* out_probs = out_att + (size_t)BATCH * HID; // [B,C]

  const size_t lds_bytes = (size_t)(A_HALFS + 2 * B_HALFS) * sizeof(_Float16); // 256KB

  convert_w_kernel<<<(N2H * HID) / (256 * 4), 256, 0, stream>>>(W_in, Wh);
  projc_kernel<<<BATCH / 64, 128, lds_bytes, stream>>>(cell_state_c, Wh, projc);
  scores_kernel<<<MROWS / 64, 128, lds_bytes, stream>>>(cell_states, Wh, projc,
                                                        atten_v, scores);
  softmax_out_kernel<<<BATCH, 256, 0, stream>>>(scores, cell_states,
                                                out_att, out_probs);
}
